// DiagnosticPlasticLinear_86955907875211
// MI455X (gfx1250) — matrix operand order repaired and checked by reading, not compiled, not run on hardware
//
#include <hip/hip_runtime.h>
#include <hip/hip_bf16.h>

typedef __attribute__((ext_vector_type(16))) __bf16 v16bf;
typedef __attribute__((ext_vector_type(8)))  __bf16 v8bf;
typedef __attribute__((ext_vector_type(8)))  float  v8f;

#define N_TOK 4096
#define D     4096
#define KSTEP 64          // K elements staged per sync round (2 WMMA k-steps)
#define LDSTR 72          // shorts per LDS tile row (64 data + 8 pad) -> 144B rows, 16B aligned
#define TILEB (128 * LDSTR)

// ---------- helpers ----------

__device__ __forceinline__ unsigned short f2bf(float f) {
  unsigned int u = __float_as_uint(f);
  u += 0x7FFFu + ((u >> 16) & 1u);   // round-to-nearest-even bf16
  return (unsigned short)(u >> 16);
}

__device__ __forceinline__ v16bf joinbf(v8bf lo, v8bf hi) {
  v16bf r;
#pragma unroll
  for (int i = 0; i < 8; ++i) { r[i] = lo[i]; r[i + 8] = hi[i]; }
  return r;
}

// Async global->LDS stage: each thread moves 64 contiguous bytes.
// INST_OFFSET is added to BOTH the LDS and global addresses (ISA 15.18.3),
// so one base covers all four 16B transfers.
__device__ __forceinline__ void stage_async(const unsigned short* __restrict__ g,
                                            int rowBase, int k0,
                                            unsigned short* ldsBuf, int tid) {
  const int row = tid >> 1;                 // 0..127
  const int col = (tid & 1) << 5;           // 0 or 32 shorts (0/64B)
  const unsigned lds_off = (unsigned)(size_t)(ldsBuf + row * LDSTR + col);
  const unsigned long long gaddr =
      (unsigned long long)(size_t)(g + (size_t)(rowBase + row) * D + k0 + col);
  asm volatile(
      "global_load_async_to_lds_b128 %0, %1, off\n\t"
      "global_load_async_to_lds_b128 %0, %1, off offset:16\n\t"
      "global_load_async_to_lds_b128 %0, %1, off offset:32\n\t"
      "global_load_async_to_lds_b128 %0, %1, off offset:48"
      :: "v"(lds_off), "v"(gaddr) : "memory");
}

__device__ __forceinline__ void wait_async0() {
  asm volatile("s_wait_asynccnt 0x0" ::: "memory");
}

// ---------- prep: bitnet quantize + trace blend -> W_eff (bf16) ----------

__global__ __launch_bounds__(256)
void prep_w_kernel(const float* __restrict__ w, const float* __restrict__ fast,
                   const float* __restrict__ slow, unsigned short* __restrict__ weff) {
  __shared__ float red[256];
  const int row = blockIdx.x;
  const int tid = threadIdx.x;
  const float* wr = w + (size_t)row * D;
  float s = 0.f;
  for (int e = tid; e < D; e += 256) s += fabsf(wr[e]);
  red[tid] = s; __syncthreads();
  for (int k = 128; k > 0; k >>= 1) { if (tid < k) red[tid] += red[tid + k]; __syncthreads(); }
  float scale = fmaxf(red[0] * (1.0f / D), 1e-5f);
  const float inv = 1.0f / scale;
  const float* fr = fast + (size_t)row * D;
  const float* sr = slow + (size_t)row * D;
  unsigned short* orow = weff + (size_t)row * D;
  for (int e = tid; e < D; e += 256) {
    float q = fminf(fmaxf(rintf(wr[e] * inv), -1.f), 1.f);
    orow[e] = f2bf(q * scale + 0.1f * fr[e] + 0.05f * sr[e]);
  }
}

// ---------- prep: x -> bf16 (row major) and x^T -> bf16 ----------

__global__ __launch_bounds__(256)
void prep_x_kernel(const float* __restrict__ x, unsigned short* __restrict__ xbf,
                   unsigned short* __restrict__ xTbf) {
  __shared__ unsigned short tile[32][33];
  const int tid = threadIdx.x;
  const int bx = blockIdx.x, by = blockIdx.y;
  const int c = tid & 31, r0 = tid >> 5;
#pragma unroll
  for (int j = 0; j < 4; ++j) {
    int r = r0 + j * 8;
    int gr = by * 32 + r, gc = bx * 32 + c;
    unsigned short b = f2bf(x[(size_t)gr * D + gc]);
    xbf[(size_t)gr * D + gc] = b;
    tile[r][c] = b;
  }
  __syncthreads();
#pragma unroll
  for (int j = 0; j < 4; ++j) {
    int r = r0 + j * 8;
    int gr = bx * 32 + r, gc = by * 32 + c;
    xTbf[(size_t)gr * D + gc] = tile[c][r];
  }
}

// ---------- shared NT WMMA mainloop, async double-buffered ----------
// C[128x128] += A[128xK] * B[128xK]^T ; gA rows are M, gB rows are N (K contiguous).
// ldsA/ldsB each hold two 128xKSTEP tiles (double buffer).

__device__ __forceinline__ void wmma_mainloop_nt(const unsigned short* __restrict__ gA,
                                                 const unsigned short* __restrict__ gB,
                                                 int rowA, int rowB, int tid,
                                                 unsigned short* ldsA, unsigned short* ldsB,
                                                 v8f acc[4][2]) {
  const int lane = tid & 31, wid = tid >> 5;
  const int wm = wid >> 2, wn = wid & 3;
  const int arow = wm * 64 + (lane & 15);   // A frag row within tile
  const int khalf = lane >> 4;              // ISA A-layout K interleave
  const int bcol = wn * 32 + (lane & 15);   // B frag column within tile
  const int kb = khalf * 8;                 // B-layout matches A: K interleave of 8

  stage_async(gA, rowA, 0, ldsA, tid);
  stage_async(gB, rowB, 0, ldsB, tid);

  const int NIT = D / KSTEP;                // 64 iterations
  for (int it = 0; it < NIT; ++it) {
    wait_async0();        // this wave's stage of buf[it&1] is in LDS
    __syncthreads();      // all waves staged cur buf; all prior reads of next buf done
    if (it + 1 < NIT) {
      unsigned short* nA = ldsA + ((it + 1) & 1) * TILEB;
      unsigned short* nB = ldsB + ((it + 1) & 1) * TILEB;
      stage_async(gA, rowA, (it + 1) * KSTEP, nA, tid);   // overlaps with compute below
      stage_async(gB, rowB, (it + 1) * KSTEP, nB, tid);
    }
    const unsigned short* cA = ldsA + (it & 1) * TILEB;
    const unsigned short* cB = ldsB + (it & 1) * TILEB;
#pragma unroll
    for (int ks = 0; ks < 2; ++ks) {
      v16bf bfrag[2];
#pragma unroll
      for (int nt = 0; nt < 2; ++nt) {
        int co = (bcol + nt * 16) * LDSTR + ks * 32 + kb;
        bfrag[nt] = joinbf(*(const v8bf*)&cB[co], *(const v8bf*)&cB[co + 16]);
      }
#pragma unroll
      for (int mt = 0; mt < 4; ++mt) {
        int ro = (arow + mt * 16) * LDSTR + ks * 32 + khalf * 8;
        v16bf afrag = joinbf(*(const v8bf*)&cA[ro], *(const v8bf*)&cA[ro + 16]);
#pragma unroll
        for (int nt = 0; nt < 2; ++nt) {
          acc[mt][nt] = __builtin_amdgcn_wmma_f32_16x16x32_bf16(
              false, afrag, false, bfrag[nt], (short)0, acc[mt][nt], false, false);
        }
      }
    }
  }
}

// ---------- GEMM1: y = x @ W_eff^T ; emit y (f32) and relu(y)^T (bf16) ----------

__global__ __launch_bounds__(256)
void gemm1_kernel(const unsigned short* __restrict__ xbf,
                  const unsigned short* __restrict__ weff,
                  float* __restrict__ y, unsigned short* __restrict__ yaT) {
  __shared__ unsigned short ldsA[2 * TILEB];
  __shared__ unsigned short ldsB[2 * TILEB];
  const int tid = threadIdx.x;
  const int bm = blockIdx.y * 128, bn = blockIdx.x * 128;
  v8f acc[4][2];
#pragma unroll
  for (int mt = 0; mt < 4; ++mt)
#pragma unroll
    for (int nt = 0; nt < 2; ++nt)
#pragma unroll
      for (int r = 0; r < 8; ++r) acc[mt][nt][r] = 0.f;

  wmma_mainloop_nt(xbf, weff, bm, bn, tid, ldsA, ldsB, acc);

  const int lane = tid & 31, wid = tid >> 5;
  const int mbase = bm + (wid >> 2) * 64 + ((lane >> 4) << 3);
  const int nbase = bn + (wid & 3) * 32 + (lane & 15);
#pragma unroll
  for (int mt = 0; mt < 4; ++mt)
#pragma unroll
    for (int nt = 0; nt < 2; ++nt)
#pragma unroll
      for (int r = 0; r < 8; ++r) {
        int gm = mbase + mt * 16 + r;
        int gn = nbase + nt * 16;
        float v = acc[mt][nt][r];
        y[(size_t)gm * D + gn] = v;
        yaT[(size_t)gn * N_TOK + gm] = f2bf(fmaxf(v, 0.f));
      }
}

// ---------- GEMM2: fast_unclipped = 0.95*fast + 0.05/N * (relu(y)^T @ x) ----------

__global__ __launch_bounds__(256)
void gemm2_kernel(const unsigned short* __restrict__ yaT,
                  const unsigned short* __restrict__ xTbf,
                  const float* __restrict__ fast,
                  float* __restrict__ outFast, float* __restrict__ partials) {
  __shared__ unsigned short ldsA[2 * TILEB];
  __shared__ unsigned short ldsB[2 * TILEB];
  __shared__ float red[256];
  const int tid = threadIdx.x;
  const int bm = blockIdx.y * 128, bn = blockIdx.x * 128;
  v8f acc[4][2];
#pragma unroll
  for (int mt = 0; mt < 4; ++mt)
#pragma unroll
    for (int nt = 0; nt < 2; ++nt)
#pragma unroll
      for (int r = 0; r < 8; ++r) acc[mt][nt][r] = 0.f;

  wmma_mainloop_nt(yaT, xTbf, bm, bn, tid, ldsA, ldsB, acc);

  const int lane = tid & 31, wid = tid >> 5;
  const int mbase = bm + (wid >> 2) * 64 + ((lane >> 4) << 3);
  const int nbase = bn + (wid & 3) * 32 + (lane & 15);
  const float lr = 0.05f / (float)N_TOK;
  float ssq = 0.f;
#pragma unroll
  for (int mt = 0; mt < 4; ++mt)
#pragma unroll
    for (int nt = 0; nt < 2; ++nt)
#pragma unroll
      for (int r = 0; r < 8; ++r) {
        size_t idx = (size_t)(mbase + mt * 16 + r) * D + (nbase + nt * 16);
        float nf = 0.95f * fast[idx] + lr * acc[mt][nt][r];
        outFast[idx] = nf;
        ssq += nf * nf;
      }
  red[tid] = ssq; __syncthreads();
  for (int k = 128; k > 0; k >>= 1) { if (tid < k) red[tid] += red[tid + k]; __syncthreads(); }
  if (tid == 0) partials[blockIdx.y * gridDim.x + blockIdx.x] = red[0];
}

// ---------- norm reduce + homeostatic clip factor ----------

__global__ __launch_bounds__(256)
void norm_kernel(const float* __restrict__ partials, float* __restrict__ scaleOut) {
  __shared__ float red[256];
  const int tid = threadIdx.x;
  red[tid] = partials[tid] + partials[tid + 256] + partials[tid + 512] + partials[tid + 768];
  __syncthreads();
  for (int k = 128; k > 0; k >>= 1) { if (tid < k) red[tid] += red[tid + k]; __syncthreads(); }
  if (tid == 0) {
    float nrm = sqrtf(red[0]);
    scaleOut[0] = (nrm > 5.0f) ? (5.0f / (nrm + 1e-6f)) : 1.0f;
  }
}

// ---------- finalize: clip fast in place, compute slow ----------

__global__ __launch_bounds__(256)
void finalize_kernel(const float* __restrict__ slow, const float* __restrict__ scaleIn,
                     float* __restrict__ outFast, float* __restrict__ outSlow) {
  const float s = scaleIn[0];
  size_t i = ((size_t)blockIdx.x * 256 + threadIdx.x) * 4;
  float4 f = *(const float4*)(outFast + i);
  f.x *= s; f.y *= s; f.z *= s; f.w *= s;
  *(float4*)(outFast + i) = f;
  float4 sl = *(const float4*)(slow + i);
  float4 o;
  o.x = 0.99f * sl.x + 0.01f * f.x;
  o.y = 0.99f * sl.y + 0.01f * f.y;
  o.z = 0.99f * sl.z + 0.01f * f.z;
  o.w = 0.99f * sl.w + 0.01f * f.w;
  *(float4*)(outSlow + i) = o;
}

// ---------- launcher ----------

extern "C" void kernel_launch(void* const* d_in, const int* in_sizes, int n_in,
                              void* d_out, int out_size, void* d_ws, size_t ws_size,
                              hipStream_t stream) {
  (void)in_sizes; (void)n_in; (void)out_size; (void)ws_size;
  const float* x    = (const float*)d_in[0];
  const float* w    = (const float*)d_in[1];
  const float* fast = (const float*)d_in[2];
  const float* slow = (const float*)d_in[3];

  const size_t MAT = (size_t)D * D;          // 16.7M elements
  float* y       = (float*)d_out;
  float* outFast = y + MAT;
  float* outSlow = outFast + MAT;

  char* ws = (char*)d_ws;
  const size_t BF = MAT * sizeof(unsigned short);   // 32 MB per bf16 matrix
  unsigned short* xbf  = (unsigned short*)(ws);
  unsigned short* xTbf = (unsigned short*)(ws + BF);
  unsigned short* weff = (unsigned short*)(ws + 2 * BF);
  unsigned short* yaT  = (unsigned short*)(ws + 3 * BF);
  float* partials      = (float*)(ws + 4 * BF);
  float* scaleS        = partials + 1024;

  prep_w_kernel<<<D, 256, 0, stream>>>(w, fast, slow, weff);
  prep_x_kernel<<<dim3(D / 32, N_TOK / 32), 256, 0, stream>>>(x, xbf, xTbf);
  gemm1_kernel<<<dim3(32, 32), 256, 0, stream>>>(xbf, weff, y, yaT);
  gemm2_kernel<<<dim3(32, 32), 256, 0, stream>>>(yaT, xTbf, fast, outFast, partials);
  norm_kernel<<<1, 256, 0, stream>>>(partials, scaleS);
  finalize_kernel<<<(int)(MAT / 4 / 256), 256, 0, stream>>>(slow, scaleS, outFast, outSlow);
}
